// AttentionModule_29094108463811
// MI455X (gfx1250) — matrix operand order repaired and checked by reading, not compiled, not run on hardware
//
#include <hip/hip_runtime.h>
#include <hip/hip_bf16.h>
#include <math.h>

#define B_   16
#define NQ_  2048
#define NA_  2048
#define D_   1024

typedef __bf16 bf16;
typedef bf16  v16bf __attribute__((ext_vector_type(16)));
typedef bf16  v8bf  __attribute__((ext_vector_type(8)));
typedef float v8f   __attribute__((ext_vector_type(8)));

__device__ __forceinline__ v8f wmma_bf16(v16bf a, v16bf b, v8f c) {
  // 8 args: (neg_a, A, neg_b, B, c_mod, C, reuse_a, reuse_b)
  return __builtin_amdgcn_wmma_f32_16x16x32_bf16(false, a, false, b, (short)0, c,
                                                 false, false);
}

// A-matrix per-lane fragment: two 8-element runs at p and p+16
// (lanes 0-15: K = 0..7 and 16..23; lanes 16-31: +8)
__device__ __forceinline__ v16bf ld_frag_a(const bf16* p) {
  v8bf lo = *(const v8bf*)(p);
  v8bf hi = *(const v8bf*)(p + 16);
  v16bf r;
#pragma unroll
  for (int e = 0; e < 8; ++e) { r[e] = lo[e]; r[8 + e] = hi[e]; }
  return r;
}

// B-matrix per-lane fragment: two 8-element runs at p and p+16 (same K order as A)
__device__ __forceinline__ v16bf ld_frag_b(const bf16* p) {
  v8bf lo = *(const v8bf*)(p);
  v8bf hi = *(const v8bf*)(p + 16);
  v16bf r;
#pragma unroll
  for (int e = 0; e < 8; ++e) { r[e] = lo[e]; r[8 + e] = hi[e]; }
  return r;
}

// ---------------------------------------------------------------------------
// Kernel 1: f32 -> bf16 down-convert (q, a) + transposed copy qT[b][d][i].
// After this, the full working set (3 x 67 MB bf16) is L2-resident (192 MB).
// ---------------------------------------------------------------------------
__global__ void k_convert(const float* __restrict__ q, const float* __restrict__ a,
                          bf16* __restrict__ q_bf, bf16* __restrict__ a_bf,
                          bf16* __restrict__ qT) {
  size_t idx = (size_t)blockIdx.x * blockDim.x + threadIdx.x;  // exact coverage
  float qv = q[idx];
  float av = a[idx];
  bf16 qb = (bf16)qv;
  q_bf[idx] = qb;
  a_bf[idx] = (bf16)av;
  // idx = (b*NQ + i)*D + d
  size_t d  = idx & (D_ - 1);
  size_t bi = idx >> 10;
  size_t i  = bi & (NQ_ - 1);
  size_t b  = bi >> 11;
  qT[(b * (size_t)D_ + d) * NQ_ + i] = qb;
}

// ---------------------------------------------------------------------------
// Kernel 2: per a-row j: m_j = max_i <a_j, q_i>, l_j = sum_i exp(s - m_j).
// Register-blocked 2 j-tiles x 4 i-tiles per wave: each k-step loads
// 2 A-frags + 4 B-frags (12 x b128) and feeds 8 WMMAs (1.5 loads/WMMA).
// ---------------------------------------------------------------------------
__global__ void __launch_bounds__(128) k_stats(const bf16* __restrict__ a_bf,
                                               const bf16* __restrict__ q_bf,
                                               float* __restrict__ m_out,
                                               float* __restrict__ l_out) {
  const int wave = threadIdx.x >> 5;
  const int lane = threadIdx.x & 31;
  const int b  = blockIdx.x >> 4;                     // NA/128 = 16 tiles/batch
  const int j0 = ((blockIdx.x & 15) << 7) + (wave << 5);   // 32 rows per wave
  const bf16* abase = a_bf + (size_t)b * NA_ * D_;
  const bf16* qbase = q_bf + (size_t)b * NQ_ * D_;
  const int mrow = lane & 15;
  const int akb  = (lane < 16) ? 0 : 8;
  const int bkb  = (lane < 16) ? 0 : 8;
  const bf16* arow0 = abase + (size_t)(j0 + mrow) * D_ + akb;
  const bf16* arow1 = arow0 + (size_t)16 * D_;

  v8f m_run[2], l_run[2];
#pragma unroll
  for (int jt = 0; jt < 2; ++jt)
#pragma unroll
    for (int v = 0; v < 8; ++v) { m_run[jt][v] = -__builtin_inff(); l_run[jt][v] = 0.0f; }

  for (int i0 = 0; i0 < NQ_; i0 += 64) {
    const bf16* brow = qbase + (size_t)(i0 + mrow) * D_ + bkb;
    __builtin_prefetch(qbase + (size_t)(i0 + 64 + mrow) * D_, 0, 1);
    v8f acc[2][4];
#pragma unroll
    for (int jt = 0; jt < 2; ++jt)
#pragma unroll
      for (int it = 0; it < 4; ++it) acc[jt][it] = (v8f){};
#pragma unroll 2
    for (int k0 = 0; k0 < D_; k0 += 32) {
      v16bf af0 = ld_frag_a(arow0 + k0);
      v16bf af1 = ld_frag_a(arow1 + k0);
#pragma unroll
      for (int it = 0; it < 4; ++it) {
        v16bf bfr = ld_frag_b(brow + (size_t)(it * 16) * D_ + k0);
        acc[0][it] = wmma_bf16(af0, bfr, acc[0][it]);
        acc[1][it] = wmma_bf16(af1, bfr, acc[1][it]);
      }
    }
    // lane holds s at (j = j0 + jt*16 + v + 8*(lane>=16), i = i0 + it*16 + (lane&15))
#pragma unroll
    for (int jt = 0; jt < 2; ++jt)
#pragma unroll
      for (int it = 0; it < 4; ++it)
#pragma unroll
        for (int v = 0; v < 8; ++v) {
          float s  = acc[jt][it][v];
          float mo = m_run[jt][v];
          float mn = fmaxf(mo, s);
          l_run[jt][v] = l_run[jt][v] * __expf(mo - mn) + __expf(s - mn);
          m_run[jt][v] = mn;
        }
  }
  // merge the 16 i-residue lanes inside each half-wave (halves hold disjoint j)
#pragma unroll
  for (int mask = 1; mask <= 8; mask <<= 1)
#pragma unroll
    for (int jt = 0; jt < 2; ++jt)
#pragma unroll
      for (int v = 0; v < 8; ++v) {
        float mo = __shfl_xor(m_run[jt][v], mask, 32);
        float lo = __shfl_xor(l_run[jt][v], mask, 32);
        float mn = fmaxf(m_run[jt][v], mo);
        l_run[jt][v] = l_run[jt][v] * __expf(m_run[jt][v] - mn) + lo * __expf(mo - mn);
        m_run[jt][v] = mn;
      }
  if ((lane & 15) == 0) {
#pragma unroll
    for (int jt = 0; jt < 2; ++jt) {
      int jbase = j0 + (jt << 4) + ((lane >> 4) << 3);
#pragma unroll
      for (int v = 0; v < 8; ++v) {
        m_out[(size_t)b * NA_ + jbase + v] = m_run[jt][v];
        l_out[(size_t)b * NA_ + jbase + v] = l_run[jt][v];
      }
    }
  }
}

// ---------------------------------------------------------------------------
// Kernel 3: H[j, :] = sum_i P[j,i] * q[i, :],  P = exp(S - m)/l.
// Block = (b, 32-row j-tile), 8 waves. Per 32-wide i-chunk:
//   waves split S as (j-subtile x i-half x K-half), ds_add_f32 reduce in LDS,
//   P (32x32 bf16) built once in LDS, each wave PV-accumulates a 32x128 f32
//   tile (16 x v8f); each qT B-fragment feeds two WMMAs.
// ---------------------------------------------------------------------------
__global__ void __launch_bounds__(256) k_out(const bf16* __restrict__ a_bf,
                                             const bf16* __restrict__ q_bf,
                                             const bf16* __restrict__ qT,
                                             const float* __restrict__ m_in,
                                             const float* __restrict__ l_in,
                                             float* __restrict__ out) {
  __shared__ __align__(16) float Sld[32 * 32];
  __shared__ __align__(16) bf16  Pld[32 * 32];
  __shared__ float ms[32];
  __shared__ float li[32];
  const int tid  = threadIdx.x;
  const int wave = tid >> 5;
  const int lane = tid & 31;
  const int b  = blockIdx.x >> 6;                    // NA/32 = 64 tiles/batch
  const int j0 = (blockIdx.x & 63) << 5;

  if (tid < 32) {
    ms[tid] = m_in[(size_t)b * NA_ + j0 + tid];
    li[tid] = 1.0f / l_in[(size_t)b * NA_ + j0 + tid];
  }

  const bf16* abase = a_bf + (size_t)b * NA_ * D_;
  const bf16* qbase = q_bf + (size_t)b * NQ_ * D_;
  const bf16* tbase = qT   + (size_t)b * D_ * NQ_;

  const int js = wave & 1;        // j-subtile (16 of 32)
  const int ih = (wave >> 1) & 1; // i-half (16 of 32)
  const int kh = wave >> 2;       // K half (512 of 1024)
  const int d0 = wave << 7;       // 128-wide d slice per wave
  const int mrow  = lane & 15;
  const int akb   = (lane < 16) ? 0 : 8;
  const int bkb   = (lane < 16) ? 0 : 8;
  const int icol  = (ih << 4) + mrow;
  const int jrow0 = (lane >> 4) << 3;

  v8f H[2][8] = {};

  const bf16* arow = abase + (size_t)(j0 + (js << 4) + mrow) * D_ + (kh << 9) + akb;

  for (int i0 = 0; i0 < NQ_; i0 += 32) {
    Sld[tid]       = 0.0f;
    Sld[tid + 256] = 0.0f;
    Sld[tid + 512] = 0.0f;
    Sld[tid + 768] = 0.0f;
    __syncthreads();

    // partial S: this wave covers (j-subtile js, i-half ih, K-half kh): 16 WMMAs
    const bf16* brow = qbase + (size_t)(i0 + (ih << 4) + mrow) * D_ + (kh << 9) + bkb;
    v8f acc = {};
#pragma unroll
    for (int ks = 0; ks < 16; ++ks) {
      v16bf af  = ld_frag_a(arow + (ks << 5));
      v16bf bfr = ld_frag_b(brow + (ks << 5));
      acc = wmma_bf16(af, bfr, acc);
    }
#pragma unroll
    for (int v = 0; v < 8; ++v)
      atomicAdd(&Sld[((js << 4) + jrow0 + v) * 32 + icol], acc[v]);  // ds_add_f32
    __syncthreads();

    // P = exp(S - m) * (1/l), stored row-major [32][32] as bf16
    {
      int e0 = tid << 2;
#pragma unroll
      for (int u = 0; u < 4; ++u) {
        int e = e0 + u;
        int j = e >> 5;
        Pld[e] = (bf16)(__expf(Sld[e] - ms[j]) * li[j]);
      }
    }
    __syncthreads();

    // PV: H[32 x 128] += P(32x32) x Q(32x16 per d-subtile); each qT B-frag
    // is shared by the two j-subtile WMMAs.
    v16bf pf0 = ld_frag_a(&Pld[mrow * 32 + akb]);          // ds_load_b128 x2
    v16bf pf1 = ld_frag_a(&Pld[(16 + mrow) * 32 + akb]);
#pragma unroll
    for (int t = 0; t < 8; ++t) {
      const bf16* trow = tbase + (size_t)(d0 + (t << 4) + mrow) * NQ_ + i0 + bkb;
      v16bf bfr = ld_frag_b(trow);
      H[0][t] = wmma_bf16(pf0, bfr, H[0][t]);
      H[1][t] = wmma_bf16(pf1, bfr, H[1][t]);
    }
    __syncthreads();
  }

  // store: lane holds H[g][t] at (j = g*16 + jrow0 + v, d = d0 + t*16 + mrow)
#pragma unroll
  for (int g = 0; g < 2; ++g)
#pragma unroll
    for (int t = 0; t < 8; ++t) {
      float* orow = out + ((size_t)b * NA_ + j0 + (g << 4)) * D_ + d0 + (t << 4) + mrow;
#pragma unroll
      for (int v = 0; v < 8; ++v)
        orow[(size_t)(jrow0 + v) * D_] = H[g][t][v];
    }
}

// ---------------------------------------------------------------------------
extern "C" void kernel_launch(void* const* d_in, const int* in_sizes, int n_in,
                              void* d_out, int out_size, void* d_ws, size_t ws_size,
                              hipStream_t stream) {
  (void)in_sizes; (void)n_in; (void)out_size; (void)ws_size;
  const float* q = (const float*)d_in[0];
  const float* a = (const float*)d_in[1];
  float* out = (float*)d_out;

  const size_t S1 = (size_t)B_ * NQ_ * D_;   // 33,554,432 elements
  bf16*  q_bf = (bf16*)d_ws;                 // 67 MB
  bf16*  a_bf = q_bf + S1;                   // 67 MB
  bf16*  qT   = a_bf + S1;                   // 67 MB  [b][d][i]
  float* m    = (float*)(qT + S1);           // 128 KB
  float* l    = m + (size_t)B_ * NA_;        // 128 KB

  k_convert<<<(unsigned)(S1 / 256), 256, 0, stream>>>(q, a, q_bf, a_bf, qT);
  k_stats  <<<B_ * (NA_ / 128), 128, 0, stream>>>(a_bf, q_bf, m, l);
  k_out    <<<B_ * (NA_ / 32), 256, 0, stream>>>(a_bf, q_bf, qT, m, l, out);
}
